// StereoPoseNet_with_depth_for_realdemo_82222853915019
// MI455X (gfx1250) — matrix operand order repaired and checked by reading, not compiled, not run on hardware
//
#include <hip/hip_runtime.h>
#include <math.h>

#define DEV __device__ __forceinline__

typedef __attribute__((ext_vector_type(16))) __bf16 bf16x16;
typedef __attribute__((ext_vector_type(8)))  float  floatx8;

namespace {
constexpr int IMG_H = 96, IMG_W = 96, HW = IMG_H * IMG_W;   // 9216
constexpr int NDEPTH = 32, NPOINTS = 1024, FEATC = 32;
constexpr int VOXF = NDEPTH * HW;                            // 294912 full-res voxels
constexpr int ATAB_ENT = 28;                                 // 27 taps + zero-pad slot
constexpr int ATAB_BYTES = 128 * ATAB_ENT * 4;               // 14336 B (16B aligned)
}

// ---------------- bf16 helpers (round-to-nearest-even) ----------------
DEV unsigned short f2bf(float f) {
  union { float f; unsigned u; } v; v.f = f;
  unsigned u = v.u;
  u += 0x7FFFu + ((u >> 16) & 1u);
  return (unsigned short)(u >> 16);
}
DEV float bf2f(unsigned short h) {
  union { unsigned u; float f; } v; v.u = ((unsigned)h) << 16; return v.f;
}

// ---------------- projection setup: rot/trans for both warp directions ----
DEV void inv4x4(const float* A, float* out) {
  float m[4][8];
  for (int r = 0; r < 4; ++r)
    for (int c = 0; c < 4; ++c) { m[r][c] = A[r*4+c]; m[r][c+4] = (r == c) ? 1.f : 0.f; }
  for (int col = 0; col < 4; ++col) {
    int piv = col;
    for (int r = col+1; r < 4; ++r) if (fabsf(m[r][col]) > fabsf(m[piv][col])) piv = r;
    if (piv != col) for (int c = 0; c < 8; ++c) { float t = m[col][c]; m[col][c] = m[piv][c]; m[piv][c] = t; }
    float inv = 1.f / m[col][col];
    for (int c = 0; c < 8; ++c) m[col][c] *= inv;
    for (int r = 0; r < 4; ++r) if (r != col) {
      float f = m[r][col];
      for (int c = 0; c < 8; ++c) m[r][c] -= f * m[col][c];
    }
  }
  for (int r = 0; r < 4; ++r) for (int c = 0; c < 4; ++c) out[r*4+c] = m[r][c+4];
}
DEV void mat4mul(const float* A, const float* B, float* C) {
  for (int r = 0; r < 4; ++r)
    for (int c = 0; c < 4; ++c) {
      float s = 0.f;
      for (int k = 0; k < 4; ++k) s += A[r*4+k] * B[k*4+c];
      C[r*4+c] = s;
    }
}
__global__ void k_setup_rt(const float* __restrict__ p1, const float* __restrict__ p2,
                           float* __restrict__ rt) {
  if (threadIdx.x != 0 || blockIdx.x != 0) return;
  float i1[16], i2[16], P[16];
  inv4x4(p1, i1); inv4x4(p2, i2);
  mat4mul(p2, i1, P);   // warp f2 into view1 frustum
  for (int r = 0; r < 3; ++r) { for (int c = 0; c < 3; ++c) rt[r*3+c] = P[r*4+c]; rt[9+r] = P[r*4+3]; }
  mat4mul(p1, i2, P);   // warp f1 into view2 frustum
  for (int r = 0; r < 3; ++r) { for (int c = 0; c < 3; ++c) rt[12+r*3+c] = P[r*4+c]; rt[21+r] = P[r*4+3]; }
}

// ---- 256B zero page used as the A-operand pad target ---------------------
__global__ void k_zeroinit(unsigned int* __restrict__ z) {
  z[threadIdx.x] = 0u;   // launched with 64 threads -> 256 bytes
}

// ---------------- direct 2D conv 3x3 pad1 + bias + relu (f32) ------------
__global__ __launch_bounds__(256)
void k_conv2d(const float* __restrict__ in, const float* __restrict__ w,
              const float* __restrict__ bias, float* __restrict__ out,
              int inC, int outC) {
  int idx = blockIdx.x * 256 + threadIdx.x;
  if (idx >= outC * HW) return;
  int oc = idx / HW, p = idx % HW;
  int y = p / IMG_W, x = p % IMG_W;
  float acc = bias[oc];
  for (int ic = 0; ic < inC; ++ic) {
    const float* ip = in + ic * HW;
    const float* wp = w + (oc * inC + ic) * 9;
    for (int ky = 0; ky < 3; ++ky) {
      int iy = y + ky - 1;
      if (iy < 0 || iy >= IMG_H) continue;
      for (int kx = 0; kx < 3; ++kx) {
        int ix = x + kx - 1;
        if (ix < 0 || ix >= IMG_W) continue;
        acc += ip[iy * IMG_W + ix] * wp[ky * 3 + kx];
      }
    }
  }
  out[idx] = acc > 0.f ? acc : 0.f;
}

// ---- homography warp + bilinear sample + variance cost (= -2*ref*warp) ---
__global__ __launch_bounds__(256)
void k_warp_var(const float* __restrict__ fref, const float* __restrict__ fsrc,
                const float* __restrict__ rt, const float* __restrict__ depthv,
                unsigned short* __restrict__ cv) {
  int idx = blockIdx.x * 256 + threadIdx.x;
  if (idx >= VOXF) return;
  int pix = idx % HW;
  int d   = idx / HW;
  float X = (float)(pix % IMG_W), Y = (float)(pix / IMG_W);
  float dep = depthv[d];
  float cx = (rt[0]*X + rt[1]*Y + rt[2]) * dep + rt[9];
  float cy = (rt[3]*X + rt[4]*Y + rt[5]) * dep + rt[10];
  float cz = (rt[6]*X + rt[7]*Y + rt[8]) * dep + rt[11];
  float xn = (cx / cz) / ((IMG_W - 1) * 0.5f) - 1.f;
  float yn = (cy / cz) / ((IMG_H - 1) * 0.5f) - 1.f;
  float xs = ((xn + 1.f) * IMG_W - 1.f) * 0.5f;
  float ys = ((yn + 1.f) * IMG_H - 1.f) * 0.5f;
  float x0f = floorf(xs), y0f = floorf(ys);
  float wx = xs - x0f, wy = ys - y0f;
  int x0 = (int)x0f, y0 = (int)y0f;
  int x1 = x0 + 1, y1 = y0 + 1;
  auto clampi = [](int v, int lo, int hi) { return v < lo ? lo : (v > hi ? hi : v); };
  int xc0 = clampi(x0, 0, IMG_W-1), xc1 = clampi(x1, 0, IMG_W-1);
  int yc0 = clampi(y0, 0, IMG_H-1), yc1 = clampi(y1, 0, IMG_H-1);
  float vx0 = (x0 >= 0 && x0 < IMG_W) ? 1.f : 0.f;
  float vx1 = (x1 >= 0 && x1 < IMG_W) ? 1.f : 0.f;
  float vy0 = (y0 >= 0 && y0 < IMG_H) ? 1.f : 0.f;
  float vy1 = (y1 >= 0 && y1 < IMG_H) ? 1.f : 0.f;
  float w00 = (1.f-wx)*(1.f-wy)*vx0*vy0, w10 = wx*(1.f-wy)*vx1*vy0;
  float w01 = (1.f-wx)*wy*vx0*vy1,       w11 = wx*wy*vx1*vy1;
  int i00 = yc0*IMG_W+xc0, i10 = yc0*IMG_W+xc1, i01 = yc1*IMG_W+xc0, i11 = yc1*IMG_W+xc1;
  unsigned short* o = cv + (size_t)idx * FEATC;
  for (int c = 0; c < FEATC; ++c) {
    const float* sp = fsrc + c * HW;
    float s = w00*sp[i00] + w10*sp[i10] + w01*sp[i01] + w11*sp[i11];
    o[c] = f2bf(-2.f * fref[c * HW + pix] * s);  // r^2+w^2-(r+w)^2 == -2rw
  }
}

// ---- pack conv3d weight (O,I,3,3,3) f32 -> bf16 lane-ready layout --------
// Packed: wt[((chunk*2 + hi)*outCpad + n)*16 + e] = W[k][n], k = chunk*32+hi*16+e
// where W[k][n] flattens k = tap*inC + c (tap-major).
__global__ __launch_bounds__(256)
void k_pack_w3d(const float* __restrict__ w, unsigned short* __restrict__ wt,
                int inC, int outC, int Kpad, int outCpad) {
  int idx = blockIdx.x * 256 + threadIdx.x;
  if (idx >= Kpad * outCpad) return;
  int e = idx & 15;
  int rest = idx >> 4;
  int n = rest % outCpad;
  int rest2 = rest / outCpad;
  int hi = rest2 & 1, chunk = rest2 >> 1;
  int k = chunk * 32 + hi * 16 + e;
  unsigned short val = 0;
  if (k < 27 * inC && n < outC) {
    int tap = k / inC, c = k % inC;
    val = f2bf(w[(n * inC + c) * 27 + tap]);
  }
  wt[idx] = val;
}

// ---- pack MLP weight (N,K) f32 -> bf16 lane-ready layout -----------------
__global__ __launch_bounds__(256)
void k_pack_w2d(const float* __restrict__ w, unsigned short* __restrict__ wt,
                int K, int N, int Npad) {
  int idx = blockIdx.x * 256 + threadIdx.x;
  if (idx >= K * Npad) return;
  int e = idx & 15;
  int rest = idx >> 4;
  int n = rest % Npad;
  int rest2 = rest / Npad;
  int hi = rest2 & 1, chunk = rest2 >> 1;
  int k = chunk * 32 + hi * 16 + e;
  wt[idx] = (k < K && n < N) ? f2bf(w[n * K + k]) : (unsigned short)0;
}

// ---- 3D conv as implicit GEMM on v_wmma_f32_16x16x32_bf16 ----------------
// Branch-free inner loop: per-lane tap tables hold 32-bit byte offsets
// RELATIVE TO `in` (keeps global addrspace -> global_load_b128); K-padding
// maps naturally to table slot 27 (= zero page), so no selects/branches.
// Full B n-tile prestaged in LDS.  Dynamic LDS: ATAB_BYTES + Kpad*32 bytes.
__global__ __launch_bounds__(256)
void k_conv3d_wmma(const unsigned short* __restrict__ in,
                   unsigned short* __restrict__ out,
                   const unsigned short* __restrict__ wtp,
                   const float* __restrict__ bias,
                   const unsigned short* __restrict__ zeropad,
                   int inD, int inH, int inW,
                   int outD, int outH, int outW,
                   int inC, int inCLog2, int outC, int outCpad,
                   int Kpad, int stride) {
  extern __shared__ char smem[];
  int* atab = (int*)smem;                                        // [128][28]
  unsigned short* btile = (unsigned short*)(smem + ATAB_BYTES);  // [chunk][hi][16][16]

  const int t = threadIdx.x;
  const int lane = t & 31;
  const int wave = t >> 5;
  const int outVox = outD * outH * outW;
  const int tile0 = blockIdx.x * 8;                 // 8 waves per block
  const int nbase = blockIdx.y * 16;
  const int li = lane & 15, hi = lane >> 4;

  // ---- prologue A: stage ALL B chunks of this n-tile into LDS ----
  {
    const unsigned int* wtp32 = (const unsigned int*)wtp;
    unsigned int* b32 = (unsigned int*)btile;
    const int totalDw = Kpad * 8;                   // Kpad*16 bf16 / 2
    for (int i = t; i < totalDw; i += 256) {
      int e2 = i & 7;
      int nL = (i >> 3) & 15;
      int hh = (i >> 7) & 1;
      int ch = i >> 8;
      b32[i] = wtp32[((size_t)(ch * 2 + hh) * outCpad + nbase + nL) * 8 + e2];
    }
  }
  // ---- prologue B: per-A-row tap offset table (128 rows x 28 slots) ----
  if (t < 128) {
    int wv = t >> 4, lrow = t & 15;
    int tl = tile0 + wv;
    int vb = (tl * 16 < outVox) ? tl * 16 : 0;      // clamp inactive waves
    int v = vb + lrow;
    int ox = v % outW;
    int oy = (v / outW) % outH;
    int oz = v / (outW * outH);
    int bz = oz * stride - 1, by = oy * stride - 1, bx = ox * stride - 1;
    const int zOff = (int)((const char*)zeropad - (const char*)in);
    int tap = 0;
    for (int kz = 0; kz < 3; ++kz) {
      int iz = bz + kz;
      bool vz = (iz >= 0) && (iz < inD);
      for (int ky = 0; ky < 3; ++ky) {
        int iy = by + ky;
        bool vy = vz && (iy >= 0) && (iy < inH);
        int rowOff = (((iz * inH) + iy) * inW) * inC * 2;   // bytes
        for (int kx = 0; kx < 3; ++kx, ++tap) {
          int ix = bx + kx;
          bool ok = vy && (ix >= 0) && (ix < inW);
          atab[t * ATAB_ENT + tap] = ok ? (rowOff + ix * inC * 2) : zOff;
        }
      }
    }
    atab[t * ATAB_ENT + 27] = zOff;                 // K-padding slot
  }
  __syncthreads();

  const bool active = (tile0 + wave) * 16 < outVox; // wave-uniform
  const int vbase = active ? (tile0 + wave) * 16 : 0;
  const int rowIdx = wave * 16 + li;
  const char* inB = (const char*)in;

  floatx8 acc = {0.f, 0.f, 0.f, 0.f, 0.f, 0.f, 0.f, 0.f};
  const int nChunks = Kpad >> 5;
  for (int chunk = 0; chunk < nChunks; ++chunk) {
    const int kc = chunk << 5;
    union { bf16x16 v; uint4 q[2]; } a, b;
    {
      int k0 = kc + hi * 16;                        // run 0: 8 consecutive K
      int tap = k0 >> inCLog2;                      // ==27 in K-pad region
      int c0 = k0 & (inC - 1);
      int off = atab[rowIdx * ATAB_ENT + tap];
      a.q[0] = *(const uint4*)((const unsigned short*)(inB + off) + c0);
    }
    {
      int k0 = kc + hi * 16 + 8;                    // run 1
      int tap = k0 >> inCLog2;
      int c0 = k0 & (inC - 1);
      int off = atab[rowIdx * ATAB_ENT + tap];
      a.q[1] = *(const uint4*)((const unsigned short*)(inB + off) + c0);
    }
    const uint4* lb = (const uint4*)(btile + ((size_t)(chunk * 2 + hi) * 16 + li) * 16);
    b.q[0] = lb[0];
    b.q[1] = lb[1];
    acc = __builtin_amdgcn_wmma_f32_16x16x32_bf16(false, a.v, false, b.v,
                                                  (short)0, acc, false, false);
  }

  // D layout: N = lane&15; VGPR r holds M = r + 8*(lane>>4)
  const int n = nbase + li;
  if (active && n < outC) {
    float bv = bias[n];
    #pragma unroll
    for (int r = 0; r < 8; ++r) {
      int m = vbase + r + (hi ? 8 : 0);
      float x = acc[r] + bv;
      x = x > 0.f ? x : 0.f;
      out[(size_t)m * outC + n] = f2bf(x);
    }
  }
}

// ---- plain GEMM on WMMA for the per-point MLP head -----------------------
// A: bf16 [M][K] (K multiple of 32, rows 16B aligned); wtp lane-ready packed.
__global__ __launch_bounds__(256)
void k_gemm_wmma(const unsigned short* __restrict__ A,
                 unsigned short* __restrict__ out,
                 const unsigned short* __restrict__ wtp,
                 const float* __restrict__ bias,
                 int M, int K, int N, int Npad) {
  const int lane = threadIdx.x & 31;
  const int wave = threadIdx.x >> 5;
  const int tile = blockIdx.x * (blockDim.x >> 5) + wave;
  const int mbase = tile * 16;
  if (mbase >= M) return;                        // wave-uniform; no barriers here
  const int nbase = blockIdx.y * 16;
  const int li = lane & 15, hi = lane >> 4;
  const int row = mbase + li;
  floatx8 acc = {0.f, 0.f, 0.f, 0.f, 0.f, 0.f, 0.f, 0.f};
  const int nChunks = K >> 5;
  for (int chunk = 0; chunk < nChunks; ++chunk) {
    const int kc = chunk << 5;
    union { bf16x16 v; uint4 q[2]; } a, b;
    a.q[0] = *(const uint4*)(A + (size_t)row * K + kc + hi * 16);
    a.q[1] = *(const uint4*)(A + (size_t)row * K + kc + hi * 16 + 8);
    const uint4* wp4 =
        (const uint4*)(wtp + (((size_t)(chunk * 2 + hi)) * Npad + nbase + li) * 16);
    b.q[0] = wp4[0];
    b.q[1] = wp4[1];
    acc = __builtin_amdgcn_wmma_f32_16x16x32_bf16(false, a.v, false, b.v,
                                                  (short)0, acc, false, false);
  }
  const int n = nbase + li;
  if (n < N) {
    float bv = bias[n];
    #pragma unroll
    for (int r = 0; r < 8; ++r) {
      float x = acc[r] + bv;
      x = x > 0.f ? x : 0.f;
      out[(size_t)(mbase + r + (hi ? 8 : 0)) * N + n] = f2bf(x);
    }
  }
}

// ---- transposed 3D conv (lhs_dilation=2, pad lo=2 hi=1) + relu + skip ----
// jax conv_transpose(SAME, k=3, s=2): pad_a=2, pad_b=1; no kernel flip.
__global__ __launch_bounds__(256)
void k_deconv3d(const unsigned short* __restrict__ in,   // [inVox][inC]
                const float* __restrict__ w,             // [outC][inC][27]
                const float* __restrict__ bias,
                const unsigned short* __restrict__ skip, // [outVox][outC]
                unsigned short* __restrict__ out,        // [outVox][outC]
                int inD, int inH, int inW, int inC, int outC) {
  const int outD = inD * 2, outH = inH * 2, outW = inW * 2;
  const int total = outD * outH * outW * outC;
  int idx = blockIdx.x * 256 + threadIdx.x;
  if (idx >= total) return;
  int oc = idx % outC;
  int v  = idx / outC;
  int x = v % outW, y = (v / outW) % outH, z = v / (outW * outH);
  float acc = 0.f;
  for (int kz = 0; kz < 3; ++kz) {
    int dz = z + kz - 2;
    if (dz < 0 || (dz & 1) || dz > 2 * inD - 2) continue;
    int iz = dz >> 1;
    for (int ky = 0; ky < 3; ++ky) {
      int dy = y + ky - 2;
      if (dy < 0 || (dy & 1) || dy > 2 * inH - 2) continue;
      int iy = dy >> 1;
      for (int kx = 0; kx < 3; ++kx) {
        int dx = x + kx - 2;
        if (dx < 0 || (dx & 1) || dx > 2 * inW - 2) continue;
        int ix = dx >> 1;
        const unsigned short* ip = in + (size_t)(((iz * inH) + iy) * inW + ix) * inC;
        const float* wp = w + (size_t)(oc * inC) * 27 + (kz * 3 + ky) * 3 + kx;
        for (int i = 0; i < inC; ++i) acc += bf2f(ip[i]) * wp[(size_t)i * 27];
      }
    }
  }
  float r = acc + bias[oc];
  r = r > 0.f ? r : 0.f;
  out[idx] = f2bf(bf2f(skip[idx]) + r);
}

// ---- final conv3d 8->1 (no relu), f32 out --------------------------------
__global__ __launch_bounds__(256)
void k_pwconv(const unsigned short* __restrict__ in,  // [vox][8]
              const float* __restrict__ w,            // [8][27] (o==0)
              const float* __restrict__ bias,
              float* __restrict__ p) {
  int idx = blockIdx.x * 256 + threadIdx.x;
  if (idx >= VOXF) return;
  int x = idx % IMG_W, y = (idx / IMG_W) % IMG_H, z = idx / HW;
  float acc = bias[0];
  for (int kz = 0; kz < 3; ++kz) {
    int iz = z + kz - 1; if (iz < 0 || iz >= NDEPTH) continue;
    for (int ky = 0; ky < 3; ++ky) {
      int iy = y + ky - 1; if (iy < 0 || iy >= IMG_H) continue;
      for (int kx = 0; kx < 3; ++kx) {
        int ix = x + kx - 1; if (ix < 0 || ix >= IMG_W) continue;
        const unsigned short* ip = in + (size_t)(((iz * IMG_H) + iy) * IMG_W + ix) * 8;
        int tap = (kz * 3 + ky) * 3 + kx;
        for (int c = 0; c < 8; ++c) acc += bf2f(ip[c]) * w[c * 27 + tap];
      }
    }
  }
  p[idx] = acc;
}

// ---- gather chosen pixels -> bf16 [NPOINTS][32] --------------------------
__global__ __launch_bounds__(256)
void k_gather(const float* __restrict__ f, const int* __restrict__ choose,
              unsigned short* __restrict__ e0) {
  int idx = blockIdx.x * 256 + threadIdx.x;
  if (idx >= NPOINTS * FEATC) return;
  int j = idx / FEATC, c = idx % FEATC;
  e0[idx] = f2bf(f[c * HW + choose[j]]);
}

// ---- last MLP layer (3x64) with tanh, writes (NPTS,3) --------------------
__global__ __launch_bounds__(256)
void k_head_tanh(const unsigned short* __restrict__ e3,
                 const float* __restrict__ w, const float* __restrict__ b,
                 float* __restrict__ outp) {
  int idx = blockIdx.x * 256 + threadIdx.x;
  if (idx >= NPOINTS * 3) return;
  int j = idx / 3, o = idx % 3;
  float acc = b[o];
  for (int c = 0; c < 64; ++c) acc += bf2f(e3[j * 64 + c]) * w[o * 64 + c];
  outp[j * 3 + o] = tanhf(acc);
}

// ---- depth expectation: softmax over D of prob volume at chosen pixels ---
__global__ __launch_bounds__(256)
void k_depth(const float* __restrict__ p, const int* __restrict__ choose,
             const float* __restrict__ dv, float* __restrict__ outd) {
  int j = blockIdx.x * 256 + threadIdx.x;
  if (j >= NPOINTS) return;
  int pix = choose[j];
  float m = -3.4e38f;
  for (int d = 0; d < NDEPTH; ++d) { float v = p[d * HW + pix]; m = v > m ? v : m; }
  float s = 0.f, acc = 0.f;
  for (int d = 0; d < NDEPTH; ++d) {
    float e = expf(p[d * HW + pix] - m);
    s += e; acc += e * dv[d];
  }
  outd[j] = acc / s;
}

// -------------------------------------------------------------------------
static inline int divUp(int a, int b) { return (a + b - 1) / b; }

extern "C" void kernel_launch(void* const* d_in, const int* in_sizes, int n_in,
                              void* d_out, int out_size, void* d_ws, size_t ws_size,
                              hipStream_t stream) {
  const float* v1img  = (const float*)d_in[0];
  const float* v2img  = (const float*)d_in[1];
  const int*   v1ch   = (const int*)d_in[2];
  const int*   v2ch   = (const int*)d_in[3];
  const float* v1proj = (const float*)d_in[6];
  const float* v2proj = (const float*)d_in[7];
  const float* depthv = (const float*)d_in[8];
  const float* fe_w1 = (const float*)d_in[9],  *fe_b1 = (const float*)d_in[10];
  const float* fe_w2 = (const float*)d_in[11], *fe_b2 = (const float*)d_in[12];
  const float* ic_w  = (const float*)d_in[13], *ic_b  = (const float*)d_in[14];
  const float* nh_w1 = (const float*)d_in[15], *nh_b1 = (const float*)d_in[16];
  const float* nh_w2 = (const float*)d_in[17], *nh_b2 = (const float*)d_in[18];
  const float* nh_w3 = (const float*)d_in[19], *nh_b3 = (const float*)d_in[20];
  const float* cw0 = (const float*)d_in[21], *cb0 = (const float*)d_in[22];
  const float* cw1 = (const float*)d_in[23], *cb1 = (const float*)d_in[24];
  const float* cw2 = (const float*)d_in[25], *cb2 = (const float*)d_in[26];
  const float* cw3 = (const float*)d_in[27], *cb3 = (const float*)d_in[28];
  const float* cw4 = (const float*)d_in[29], *cb4 = (const float*)d_in[30];
  const float* cw5 = (const float*)d_in[31], *cb5 = (const float*)d_in[32];
  const float* cw6 = (const float*)d_in[33], *cb6 = (const float*)d_in[34];
  const float* dw7  = (const float*)d_in[35], *db7  = (const float*)d_in[36];
  const float* dw9  = (const float*)d_in[37], *db9  = (const float*)d_in[38];
  const float* dw11 = (const float*)d_in[39], *db11 = (const float*)d_in[40];
  const float* pwv  = (const float*)d_in[41], *pbv  = (const float*)d_in[42];
  float* outp = (float*)d_out;

  // ---- workspace bump allocator (aligned 256B) ----
  char* base = (char*)d_ws;
  size_t off = 0;
  auto alloc = [&](size_t bytes) -> void* {
    void* r = base + off;
    off = (off + bytes + 255) & ~(size_t)255;
    return r;
  };
  float* rt    = (float*)alloc(24 * sizeof(float));
  unsigned short* zpad = (unsigned short*)alloc(256);
  float* f1    = (float*)alloc((size_t)FEATC * HW * 4);
  float* f2    = (float*)alloc((size_t)FEATC * HW * 4);
  float* t2d   = (float*)alloc((size_t)FEATC * HW * 4);
  unsigned short* cv  = (unsigned short*)alloc((size_t)VOXF * 32 * 2);
  unsigned short* x0  = (unsigned short*)alloc((size_t)VOXF * 8 * 2);
  unsigned short* x1  = (unsigned short*)alloc((size_t)36864 * 16 * 2);
  unsigned short* x2  = (unsigned short*)alloc((size_t)36864 * 16 * 2);
  unsigned short* x3  = (unsigned short*)alloc((size_t)4608 * 32 * 2);
  unsigned short* x4  = (unsigned short*)alloc((size_t)4608 * 32 * 2);
  unsigned short* x5  = (unsigned short*)alloc((size_t)576 * 64 * 2);
  unsigned short* y6  = (unsigned short*)alloc((size_t)576 * 64 * 2);
  unsigned short* y7  = (unsigned short*)alloc((size_t)4608 * 32 * 2);
  unsigned short* y9  = (unsigned short*)alloc((size_t)36864 * 16 * 2);
  unsigned short* y11 = (unsigned short*)alloc((size_t)VOXF * 8 * 2);
  float* pvol = (float*)alloc((size_t)VOXF * 4);
  unsigned short* wt0 = (unsigned short*)alloc((size_t)864 * 16 * 2);
  unsigned short* wt1 = (unsigned short*)alloc((size_t)224 * 16 * 2);
  unsigned short* wt2 = (unsigned short*)alloc((size_t)448 * 16 * 2);
  unsigned short* wt3 = (unsigned short*)alloc((size_t)448 * 32 * 2);
  unsigned short* wt4 = (unsigned short*)alloc((size_t)864 * 32 * 2);
  unsigned short* wt5 = (unsigned short*)alloc((size_t)864 * 64 * 2);
  unsigned short* wt6 = (unsigned short*)alloc((size_t)1728 * 64 * 2);
  unsigned short* mw1 = (unsigned short*)alloc((size_t)32 * 64 * 2);
  unsigned short* mw2 = (unsigned short*)alloc((size_t)64 * 128 * 2);
  unsigned short* mw3 = (unsigned short*)alloc((size_t)128 * 64 * 2);
  unsigned short* e0  = (unsigned short*)alloc((size_t)NPOINTS * 32 * 2);
  unsigned short* e1  = (unsigned short*)alloc((size_t)NPOINTS * 64 * 2);
  unsigned short* e2  = (unsigned short*)alloc((size_t)NPOINTS * 128 * 2);
  unsigned short* e3  = (unsigned short*)alloc((size_t)NPOINTS * 64 * 2);
  (void)ws_size; (void)in_sizes; (void)n_in; (void)out_size;

  // ---- projection setup + zero page ----
  k_setup_rt<<<1, 32, 0, stream>>>(v1proj, v2proj, rt);
  k_zeroinit<<<1, 64, 0, stream>>>((unsigned int*)zpad);

  // ---- feature extraction (direct f32) ----
  k_conv2d<<<1152, 256, 0, stream>>>(v1img, fe_w1, fe_b1, t2d, 3, 32);
  k_conv2d<<<1152, 256, 0, stream>>>(t2d, fe_w2, fe_b2, f1, 32, 32);
  k_conv2d<<<1152, 256, 0, stream>>>(v2img, fe_w1, fe_b1, t2d, 3, 32);
  k_conv2d<<<1152, 256, 0, stream>>>(t2d, fe_w2, fe_b2, f2, 32, 32);

  // ---- pack weights to bf16 lane-ready layout once ----
  k_pack_w3d<<<divUp(864 * 16, 256), 256, 0, stream>>>(cw0, wt0, 32, 8, 864, 16);
  k_pack_w3d<<<divUp(224 * 16, 256), 256, 0, stream>>>(cw1, wt1, 8, 16, 224, 16);
  k_pack_w3d<<<divUp(448 * 16, 256), 256, 0, stream>>>(cw2, wt2, 16, 16, 448, 16);
  k_pack_w3d<<<divUp(448 * 32, 256), 256, 0, stream>>>(cw3, wt3, 16, 32, 448, 32);
  k_pack_w3d<<<divUp(864 * 32, 256), 256, 0, stream>>>(cw4, wt4, 32, 32, 864, 32);
  k_pack_w3d<<<divUp(864 * 64, 256), 256, 0, stream>>>(cw5, wt5, 32, 64, 864, 64);
  k_pack_w3d<<<divUp(1728 * 64, 256), 256, 0, stream>>>(cw6, wt6, 64, 64, 1728, 64);
  k_pack_w2d<<<divUp(32 * 64, 256), 256, 0, stream>>>(ic_w, mw1, 32, 64, 64);
  k_pack_w2d<<<divUp(64 * 128, 256), 256, 0, stream>>>(nh_w1, mw2, 64, 128, 128);
  k_pack_w2d<<<divUp(128 * 64, 256), 256, 0, stream>>>(nh_w2, mw3, 128, 64, 64);

  // dynamic LDS per conv layer: A-table + full B tile
  auto smemFor = [](int Kpad) -> size_t { return (size_t)ATAB_BYTES + (size_t)Kpad * 32; };

  // ---- per-view pipeline (volumes processed sequentially, buffers reused) --
  for (int vi = 0; vi < 2; ++vi) {
    const float* fref = (vi == 0) ? f1 : f2;
    const float* fsrc = (vi == 0) ? f2 : f1;
    const float* rtp  = rt + ((vi == 0) ? 0 : 12);
    const int* choose = (vi == 0) ? v1ch : v2ch;
    float* nOut = outp + ((vi == 0) ? 0 : 3072);
    float* dOut = outp + ((vi == 0) ? 6144 : 7168);

    // cost volume: variance(ref, warp(src)) == -2*ref*warp, bf16 channel-last
    k_warp_var<<<1152, 256, 0, stream>>>(fref, fsrc, rtp, depthv, cv);

    // 3D UNet encoder (all WMMA implicit GEMM)
    k_conv3d_wmma<<<dim3(2304, 1), 256, smemFor(864), stream>>>(cv, x0, wt0, cb0, zpad,
        32, 96, 96, 32, 96, 96, 32, 5, 8, 16, 864, 1);
    k_conv3d_wmma<<<dim3(288, 1), 256, smemFor(224), stream>>>(x0, x1, wt1, cb1, zpad,
        32, 96, 96, 16, 48, 48, 8, 3, 16, 16, 224, 2);
    k_conv3d_wmma<<<dim3(288, 1), 256, smemFor(448), stream>>>(x1, x2, wt2, cb2, zpad,
        16, 48, 48, 16, 48, 48, 16, 4, 16, 16, 448, 1);
    k_conv3d_wmma<<<dim3(36, 2), 256, smemFor(448), stream>>>(x2, x3, wt3, cb3, zpad,
        16, 48, 48, 8, 24, 24, 16, 4, 32, 32, 448, 2);
    k_conv3d_wmma<<<dim3(36, 2), 256, smemFor(864), stream>>>(x3, x4, wt4, cb4, zpad,
        8, 24, 24, 8, 24, 24, 32, 5, 32, 32, 864, 1);
    k_conv3d_wmma<<<dim3(5, 4), 256, smemFor(864), stream>>>(x4, x5, wt5, cb5, zpad,
        8, 24, 24, 4, 12, 12, 32, 5, 64, 64, 864, 2);
    k_conv3d_wmma<<<dim3(5, 4), 256, smemFor(1728), stream>>>(x5, y6, wt6, cb6, zpad,
        4, 12, 12, 4, 12, 12, 64, 6, 64, 64, 1728, 1);

    // decoder: skip + relu(deconv(.)) fused
    k_deconv3d<<<576, 256, 0, stream>>>(y6, dw7, db7, x4, y7, 4, 12, 12, 64, 32);
    k_deconv3d<<<2304, 256, 0, stream>>>(y7, dw9, db9, x2, y9, 8, 24, 24, 32, 16);
    k_deconv3d<<<9216, 256, 0, stream>>>(y9, dw11, db11, x0, y11, 16, 48, 48, 16, 8);
    k_pwconv<<<1152, 256, 0, stream>>>(y11, pwv, pbv, pvol);
    k_depth<<<4, 256, 0, stream>>>(pvol, choose, depthv, dOut);

    // NOCS head: gather + 3 WMMA GEMM layers + tanh head
    k_gather<<<128, 256, 0, stream>>>(fref, choose, e0);
    k_gemm_wmma<<<dim3(8, 4), 256, 0, stream>>>(e0, e1, mw1, ic_b, 1024, 32, 64, 64);
    k_gemm_wmma<<<dim3(8, 8), 256, 0, stream>>>(e1, e2, mw2, nh_b1, 1024, 64, 128, 128);
    k_gemm_wmma<<<dim3(8, 4), 256, 0, stream>>>(e2, e3, mw3, nh_b2, 1024, 128, 64, 64);
    k_head_tanh<<<12, 256, 0, stream>>>(e3, nh_w3, nh_b3, nOut);
  }
}
